// MultiHeadDifferentialAttention_59167469469813
// MI455X (gfx1250) — matrix operand order repaired and checked by reading, not compiled, not run on hardware
//
#include <hip/hip_runtime.h>
#include <hip/hip_bf16.h>
#include <math.h>

#define BB 4
#define NN 1024
#define EE 1024
#define HH 16
#define DD 64
#define D2 128
#define E2 2048
#define BN 4096            // B*N rows
#define SCALE 0.125f       // 1/sqrt(64)
#define LAMB_INIT 0.8f

typedef __attribute__((ext_vector_type(16))) __bf16 bf16x16;
typedef __attribute__((ext_vector_type(8)))  float  f32x8;

#define WMMA_BF16(a, b, c) \
  __builtin_amdgcn_wmma_f32_16x16x32_bf16(false, (a), false, (b), (short)0, (c), false, false)

// ---- CDNA5 async global->LDS copy (ASYNCcnt) ------------------------------
__device__ __forceinline__ void wait_async0() {
  asm volatile("s_wait_asynccnt 0x0" ::: "memory");
}

// Copy ROWS rows x KELEMS bf16 from global (row stride ld elems) into LDS at
// byte offset ldsbase, row-major [ROWS][KELEMS]. 16B per async op per lane.
template <int ROWS, int KELEMS>
__device__ __forceinline__ void fill_async(const __bf16* __restrict__ src, int ld,
                                           unsigned int ldsbase, int tid) {
  constexpr int SEGR = KELEMS / 8;        // 16-byte segments per row
  constexpr int TOT = ROWS * SEGR;
  constexpr int PT = TOT / 256;           // per-thread ops
#pragma unroll
  for (int i = 0; i < PT; ++i) {
    int s = tid + i * 256;
    int n = s / SEGR, seg = s % SEGR;
    unsigned long long g = (unsigned long long)(uintptr_t)(src + (size_t)n * ld + seg * 8);
    unsigned int l = ldsbase + (unsigned int)(n * (KELEMS * 2) + seg * 16);
    asm volatile("global_load_async_to_lds_b128 %0, %1, off"
                 :: "v"(l), "v"(g) : "memory");
  }
}

// ---- fragment loaders (ISA 7.12.2 layouts, wave32) -------------------------
__device__ __forceinline__ bf16x16 load_a_bf16(const __bf16* __restrict__ p, int ld,
                                               int m0, int k0, int lane) {
  const int m = lane & 15, h = lane >> 4;
  const __bf16* r = p + (size_t)(m0 + m) * ld + k0;
  bf16x16 a;
#pragma unroll
  for (int j = 0; j < 8; ++j) { a[j] = r[h * 8 + j]; a[8 + j] = r[16 + h * 8 + j]; }
  return a;
}

__device__ __forceinline__ bf16x16 load_a_f32cvt(const float* __restrict__ p, int ld,
                                                 int m0, int k0, int lane) {
  const int m = lane & 15, h = lane >> 4;
  const float* r = p + (size_t)(m0 + m) * ld + k0;
  bf16x16 a;
#pragma unroll
  for (int j = 0; j < 8; ++j) {
    a[j]     = (__bf16)r[h * 8 + j];
    a[8 + j] = (__bf16)r[16 + h * 8 + j];
  }
  return a;
}

// B fragment (32x16): lane holds column n = lane%16, K = (lane/16)*8 + j (j<8), 16 + (lane/16)*8 + (j-8) (j>=8).
// Source in LDS, N-major [.., KELEMS=ld].
__device__ __forceinline__ bf16x16 load_b_lds(const __bf16* sm, int ld,
                                              int n0, int k0, int lane) {
  const int n = lane & 15, h = lane >> 4;
  const __bf16* r = sm + (n0 + n) * ld + k0;
  bf16x16 b;
#pragma unroll
  for (int j = 0; j < 8; ++j) { b[j] = r[h * 8 + j]; b[8 + j] = r[16 + h * 8 + j]; }
  return b;
}

// ---- elementwise conversions ----------------------------------------------
__global__ void k_cvt_bf16(const float* __restrict__ s, __bf16* __restrict__ d, int n) {
  int i = blockIdx.x * blockDim.x + threadIdx.x;
  if (i < n) d[i] = (__bf16)s[i];
}

__global__ void k_cvt_t(const float* __restrict__ s, __bf16* __restrict__ d,
                        int rows, int cols) {
  int i = blockIdx.x * blockDim.x + threadIdx.x;
  if (i < rows * cols) {
    int r = i / cols, c = i - r * cols;
    d[(size_t)c * rows + r] = (__bf16)s[i];
  }
}

__global__ void k_lambda(const float* __restrict__ lq1, const float* __restrict__ lk1,
                         const float* __restrict__ lq2, const float* __restrict__ lk2,
                         float* __restrict__ lam) {
  int h = threadIdx.x;
  if (h < HH) {
    float s1 = 0.f, s2 = 0.f;
    for (int d = 0; d < DD; ++d) {
      s1 += lq1[h * DD + d] * lk1[h * DD + d];
      s2 += lq2[h * DD + d] * lk2[h * DD + d];
    }
    lam[h] = __expf(s1) - __expf(s2) + LAMB_INIT;
  }
}

// ---- QKV projection: async-LDS double-buffered B panel --------------------
// storeMode 0: dst[(b,h,n,dd)] (Q,K)    storeMode 1: dst[(b,h,dd,n)] (V^T)
__global__ void k_gemm_qkv(const __bf16* __restrict__ X, const __bf16* __restrict__ Wt,
                           __bf16* __restrict__ dst, int storeMode) {
  __shared__ __bf16 sB[2][64 * 64];
  const int tid = threadIdx.x;
  const int lane = tid & 31;
  const int wave = tid >> 5;
  const int row0 = blockIdx.y * 128 + wave * 16;
  const int col0 = blockIdx.x * 64;
  const unsigned int lb[2] = {(unsigned int)(uintptr_t)(void*)&sB[0][0],
                              (unsigned int)(uintptr_t)(void*)&sB[1][0]};
  const __bf16* Wp = Wt + (size_t)col0 * EE;
  f32x8 c[4] = {};
  constexpr int NC = EE / 64;
  fill_async<64, 64>(Wp, EE, lb[0], tid);
  for (int kc = 0; kc < NC; ++kc) {
    wait_async0();
    __syncthreads();
    if (kc + 1 < NC) fill_async<64, 64>(Wp + (kc + 1) * 64, EE, lb[(kc + 1) & 1], tid);
    const __bf16* smB = sB[kc & 1];
#pragma unroll
    for (int ks = 0; ks < 2; ++ks) {
      bf16x16 a = load_a_bf16(X, EE, row0, kc * 64 + ks * 32, lane);
#pragma unroll
      for (int t = 0; t < 4; ++t)
        c[t] = WMMA_BF16(a, load_b_lds(smB, 64, t * 16, ks * 32, lane), c[t]);
    }
  }
  const int nidx = lane & 15, rbase = (lane >> 4) << 3;
#pragma unroll
  for (int t = 0; t < 4; ++t)
#pragma unroll
    for (int r = 0; r < 8; ++r) {
      int row = row0 + rbase + r;           // b*N + n
      int col = col0 + t * 16 + nidx;       // h*2d + dd
      int b = row >> 10, n = row & (NN - 1);
      int h = col >> 7, dd = col & (D2 - 1);
      __bf16 v = (__bf16)c[t][r];
      if (storeMode == 0)
        dst[(((size_t)(b * HH + h) * NN) + n) * D2 + dd] = v;
      else
        dst[(((size_t)(b * HH + h) * D2) + dd) * NN + n] = v;
    }
}

// ---- pass 1: per-row softmax stats for S1 and S2 --------------------------
__global__ void k_stats(const __bf16* __restrict__ Q, const __bf16* __restrict__ K,
                        float* __restrict__ mx1, float* __restrict__ sm1,
                        float* __restrict__ mx2, float* __restrict__ sm2) {
  __shared__ __bf16 sK[2][64 * D2];
  const int tid = threadIdx.x;
  const int lane = tid & 31;
  const int wave = tid >> 5;
  const int bh = blockIdx.x;
  const int row0 = blockIdx.y * 128 + wave * 16;
  const __bf16* Qb = Q + (size_t)bh * NN * D2;
  const __bf16* Kb = K + (size_t)bh * NN * D2;
  const unsigned int lb[2] = {(unsigned int)(uintptr_t)(void*)&sK[0][0],
                              (unsigned int)(uintptr_t)(void*)&sK[1][0]};
  bf16x16 qa[2][2];
#pragma unroll
  for (int s = 0; s < 2; ++s) {
    qa[s][0] = load_a_bf16(Qb, D2, row0, s * DD, lane);
    qa[s][1] = load_a_bf16(Qb, D2, row0, s * DD + 32, lane);
  }
  float mx[2][8], sume[2][8];
#pragma unroll
  for (int s = 0; s < 2; ++s)
#pragma unroll
    for (int r = 0; r < 8; ++r) { mx[s][r] = -1e30f; sume[s][r] = 0.f; }

  constexpr int NC = NN / 64;
  fill_async<64, D2>(Kb, D2, lb[0], tid);
  for (int ch = 0; ch < NC; ++ch) {
    wait_async0();
    __syncthreads();
    if (ch + 1 < NC)
      fill_async<64, D2>(Kb + (size_t)(ch + 1) * 64 * D2, D2, lb[(ch + 1) & 1], tid);
    const __bf16* smK = sK[ch & 1];
#pragma unroll
    for (int ntl = 0; ntl < 4; ++ntl) {
#pragma unroll
      for (int s = 0; s < 2; ++s) {
        f32x8 c = {};
        c = WMMA_BF16(qa[s][0], load_b_lds(smK, D2, ntl * 16, s * DD, lane), c);
        c = WMMA_BF16(qa[s][1], load_b_lds(smK, D2, ntl * 16, s * DD + 32, lane), c);
#pragma unroll
        for (int r = 0; r < 8; ++r) {
          float v = SCALE * c[r];
          float m2 = fmaxf(mx[s][r], v);
          sume[s][r] = sume[s][r] * __expf(mx[s][r] - m2) + __expf(v - m2);
          mx[s][r] = m2;
        }
      }
    }
  }
#pragma unroll
  for (int s = 0; s < 2; ++s)
#pragma unroll
    for (int r = 0; r < 8; ++r) {
      float m = mx[s][r], l = sume[s][r];
      for (int off = 1; off < 16; off <<= 1) {
        float om = __shfl_xor(m, off, 32);
        float ol = __shfl_xor(l, off, 32);
        float m2 = fmaxf(m, om);
        l = l * __expf(m - m2) + ol * __expf(om - m2);
        m = m2;
      }
      if ((lane & 15) == 0) {
        int row = row0 + ((lane >> 4) << 3) + r;
        float* M = s ? mx2 : mx1;
        float* L = s ? sm2 : sm1;
        M[bh * NN + row] = m;
        L[bh * NN + row] = l;
      }
    }
}

// ---- pass 2: recompute scores, attn = a1 - lam*a2 (fp32 out + bf16 copy) --
__global__ void k_attn(const __bf16* __restrict__ Q, const __bf16* __restrict__ K,
                       const float* __restrict__ mx1, const float* __restrict__ sm1,
                       const float* __restrict__ mx2, const float* __restrict__ sm2,
                       const float* __restrict__ lam, float* __restrict__ attn,
                       __bf16* __restrict__ attnb) {
  __shared__ __bf16 sK[2][64 * D2];
  const int tid = threadIdx.x;
  const int lane = tid & 31;
  const int wave = tid >> 5;
  const int bh = blockIdx.x;
  const int row0 = blockIdx.y * 128 + wave * 16;
  const __bf16* Qb = Q + (size_t)bh * NN * D2;
  const __bf16* Kb = K + (size_t)bh * NN * D2;
  const unsigned int lb[2] = {(unsigned int)(uintptr_t)(void*)&sK[0][0],
                              (unsigned int)(uintptr_t)(void*)&sK[1][0]};
  const float la = lam[bh & (HH - 1)];
  float o1[8], o2[8];
#pragma unroll
  for (int r = 0; r < 8; ++r) {
    int row = bh * NN + row0 + ((lane >> 4) << 3) + r;
    o1[r] = mx1[row] + __logf(sm1[row]);
    o2[r] = mx2[row] + __logf(sm2[row]);
  }
  bf16x16 qa[2][2];
#pragma unroll
  for (int s = 0; s < 2; ++s) {
    qa[s][0] = load_a_bf16(Qb, D2, row0, s * DD, lane);
    qa[s][1] = load_a_bf16(Qb, D2, row0, s * DD + 32, lane);
  }
  float* Af = attn + (size_t)bh * NN * NN;
  __bf16* Ab = attnb ? attnb + (size_t)bh * NN * NN : (__bf16*)0;
  const int nidx = lane & 15, rbase = (lane >> 4) << 3;

  constexpr int NC = NN / 64;
  fill_async<64, D2>(Kb, D2, lb[0], tid);
  for (int ch = 0; ch < NC; ++ch) {
    wait_async0();
    __syncthreads();
    if (ch + 1 < NC)
      fill_async<64, D2>(Kb + (size_t)(ch + 1) * 64 * D2, D2, lb[(ch + 1) & 1], tid);
    const __bf16* smK = sK[ch & 1];
#pragma unroll
    for (int ntl = 0; ntl < 4; ++ntl) {
      f32x8 c1 = {}, c2 = {};
      c1 = WMMA_BF16(qa[0][0], load_b_lds(smK, D2, ntl * 16, 0, lane), c1);
      c1 = WMMA_BF16(qa[0][1], load_b_lds(smK, D2, ntl * 16, 32, lane), c1);
      c2 = WMMA_BF16(qa[1][0], load_b_lds(smK, D2, ntl * 16, 64, lane), c2);
      c2 = WMMA_BF16(qa[1][1], load_b_lds(smK, D2, ntl * 16, 96, lane), c2);
      int col = ch * 64 + ntl * 16 + nidx;
#pragma unroll
      for (int r = 0; r < 8; ++r) {
        int row = row0 + rbase + r;
        float a1v = __expf(SCALE * c1[r] - o1[r]);
        float a2v = __expf(SCALE * c2[r] - o2[r]);
        float val = a1v - la * a2v;
        Af[(size_t)row * NN + col] = val;
        if (Ab) Ab[(size_t)row * NN + col] = (__bf16)val;
      }
    }
  }
}

// ---- pass 3: O = attn @ V (V^T panel via async LDS) -----------------------
__global__ void k_pv(const float* __restrict__ attnf, const __bf16* __restrict__ attnb,
                     const __bf16* __restrict__ Vt, float* __restrict__ O) {
  __shared__ __bf16 sV[2][D2 * 64];
  const int tid = threadIdx.x;
  const int lane = tid & 31;
  const int wave = tid >> 5;
  const int bh = blockIdx.x;
  const int row0 = blockIdx.y * 128 + wave * 16;
  const float* Af = attnf + (size_t)bh * NN * NN;
  const __bf16* Ab = attnb ? attnb + (size_t)bh * NN * NN : (const __bf16*)0;
  const __bf16* Vb = Vt + (size_t)bh * D2 * NN;
  const unsigned int lb[2] = {(unsigned int)(uintptr_t)(void*)&sV[0][0],
                              (unsigned int)(uintptr_t)(void*)&sV[1][0]};
  f32x8 c[8] = {};
  constexpr int NC = NN / 64;
  fill_async<D2, 64>(Vb, NN, lb[0], tid);
  for (int ch = 0; ch < NC; ++ch) {
    wait_async0();
    __syncthreads();
    if (ch + 1 < NC) fill_async<D2, 64>(Vb + (ch + 1) * 64, NN, lb[(ch + 1) & 1], tid);
    const __bf16* smV = sV[ch & 1];
#pragma unroll
    for (int ks = 0; ks < 2; ++ks) {
      int k0 = ch * 64 + ks * 32;
      bf16x16 a = Ab ? load_a_bf16(Ab, NN, row0, k0, lane)
                     : load_a_f32cvt(Af, NN, row0, k0, lane);
#pragma unroll
      for (int t = 0; t < 8; ++t)
        c[t] = WMMA_BF16(a, load_b_lds(smV, 64, t * 16, ks * 32, lane), c[t]);
    }
  }
  float* Ob = O + (size_t)bh * NN * D2;
  const int nidx = lane & 15, rbase = (lane >> 4) << 3;
#pragma unroll
  for (int t = 0; t < 8; ++t)
#pragma unroll
    for (int r = 0; r < 8; ++r) {
      int row = row0 + rbase + r;
      Ob[(size_t)row * D2 + t * 16 + nidx] = c[t][r];
    }
}

// ---- GroupNorm over (n, 2d) per (b,h) -------------------------------------
__global__ void k_gnstats(const float* __restrict__ O, float* __restrict__ mu,
                          float* __restrict__ rsig) {
  __shared__ float ssum[256], ssq[256];
  int bh = blockIdx.x, tid = threadIdx.x;
  const float* Ob = O + (size_t)bh * NN * D2;
  float s = 0.f, q = 0.f;
  for (int i = tid; i < NN * D2; i += 256) { float v = Ob[i]; s += v; q += v * v; }
  ssum[tid] = s; ssq[tid] = q;
  __syncthreads();
  for (int st = 128; st > 0; st >>= 1) {
    if (tid < st) { ssum[tid] += ssum[tid + st]; ssq[tid] += ssq[tid + st]; }
    __syncthreads();
  }
  if (tid == 0) {
    float m = ssum[0] * (1.f / (NN * D2));
    float v = ssq[0] * (1.f / (NN * D2)) - m * m;
    mu[bh] = m;
    rsig[bh] = rsqrtf(v + 1e-5f);
  }
}

__global__ void k_gnapply(const float* __restrict__ O, const float* __restrict__ mu,
                          const float* __restrict__ rsig, const float* __restrict__ gamma,
                          const float* __restrict__ beta, __bf16* __restrict__ Y) {
  int i = blockIdx.x * 256 + threadIdx.x;           // over BH*NN*D2 = 8M
  int dd = i & (D2 - 1);
  int n  = (i >> 7) & (NN - 1);
  int bh = i >> 17;
  int b = bh >> 4, h = bh & (HH - 1);
  float y = (O[i] - mu[bh]) * rsig[bh] * gamma[dd] + beta[dd];
  Y[((size_t)(b * NN + n)) * E2 + h * D2 + dd] = (__bf16)y;
}

// ---- final projection: out = Y @ Wo (async-LDS B panel) -------------------
__global__ void k_out(const __bf16* __restrict__ Y, const __bf16* __restrict__ Wot,
                      float* __restrict__ out) {
  __shared__ __bf16 sB[2][64 * 64];
  const int tid = threadIdx.x;
  const int lane = tid & 31;
  const int wave = tid >> 5;
  const int row0 = blockIdx.y * 128 + wave * 16;
  const int col0 = blockIdx.x * 64;
  const unsigned int lb[2] = {(unsigned int)(uintptr_t)(void*)&sB[0][0],
                              (unsigned int)(uintptr_t)(void*)&sB[1][0]};
  const __bf16* Wp = Wot + (size_t)col0 * E2;
  f32x8 c[4] = {};
  constexpr int NC = E2 / 64;
  fill_async<64, 64>(Wp, E2, lb[0], tid);
  for (int kc = 0; kc < NC; ++kc) {
    wait_async0();
    __syncthreads();
    if (kc + 1 < NC) fill_async<64, 64>(Wp + (kc + 1) * 64, E2, lb[(kc + 1) & 1], tid);
    const __bf16* smB = sB[kc & 1];
#pragma unroll
    for (int ks = 0; ks < 2; ++ks) {
      bf16x16 a = load_a_bf16(Y, E2, row0, kc * 64 + ks * 32, lane);
#pragma unroll
      for (int t = 0; t < 4; ++t)
        c[t] = WMMA_BF16(a, load_b_lds(smB, 64, t * 16, ks * 32, lane), c[t]);
    }
  }
  const int nidx = lane & 15, rbase = (lane >> 4) << 3;
#pragma unroll
  for (int t = 0; t < 4; ++t)
#pragma unroll
    for (int r = 0; r < 8; ++r) {
      int row = row0 + rbase + r;
      out[(size_t)row * EE + col0 + t * 16 + nidx] = c[t][r];
    }
}

// ---------------------------------------------------------------------------
extern "C" void kernel_launch(void* const* d_in, const int* in_sizes, int n_in,
                              void* d_out, int out_size, void* d_ws, size_t ws_size,
                              hipStream_t stream) {
  (void)in_sizes; (void)n_in; (void)out_size;
  const float* x    = (const float*)d_in[0];
  const float* Wq   = (const float*)d_in[1];
  const float* Wk   = (const float*)d_in[2];
  const float* Wv   = (const float*)d_in[3];
  const float* lq1  = (const float*)d_in[4];
  const float* lk1  = (const float*)d_in[5];
  const float* lq2  = (const float*)d_in[6];
  const float* lk2  = (const float*)d_in[7];
  const float* gamma= (const float*)d_in[8];
  const float* beta = (const float*)d_in[9];
  const float* Wo   = (const float*)d_in[10];

  float* out      = (float*)d_out;
  float* attn_out = out + (size_t)BN * EE;

  char* w = (char*)d_ws;
  auto carve = [&](size_t bytes) -> void* {
    void* p = (void*)w;
    w += (bytes + 255) & ~(size_t)255;
    return p;
  };
  __bf16* xb   = (__bf16*)carve((size_t)BN * EE * 2);
  __bf16* wqt  = (__bf16*)carve((size_t)E2 * EE * 2);
  __bf16* wkt  = (__bf16*)carve((size_t)E2 * EE * 2);
  __bf16* wvt  = (__bf16*)carve((size_t)E2 * EE * 2);
  __bf16* wot  = (__bf16*)carve((size_t)EE * E2 * 2);
  __bf16* Qb   = (__bf16*)carve((size_t)BN * E2 * 2);   // (b,h,n,2d)
  __bf16* Kb   = (__bf16*)carve((size_t)BN * E2 * 2);   // (b,h,n,2d)
  __bf16* Vtb  = (__bf16*)carve((size_t)BN * E2 * 2);   // (b,h,2d,n)
  float*  mx1  = (float*)carve((size_t)BB * HH * NN * 4);
  float*  sm1  = (float*)carve((size_t)BB * HH * NN * 4);
  float*  mx2  = (float*)carve((size_t)BB * HH * NN * 4);
  float*  sm2  = (float*)carve((size_t)BB * HH * NN * 4);
  float*  lam  = (float*)carve(HH * 4);
  float*  Obuf = (float*)carve((size_t)BB * HH * NN * D2 * 4);
  float*  mu   = (float*)carve(BB * HH * 4);
  float*  rsig = (float*)carve(BB * HH * 4);
  __bf16* Yb   = (__bf16*)carve((size_t)BN * E2 * 2);   // (b,n,2e)

  // optional bf16 copy of attn (128MB) if the workspace is big enough
  size_t used = (size_t)(w - (char*)d_ws);
  size_t attnb_bytes = (size_t)BB * HH * NN * NN * 2;
  __bf16* attnb = (ws_size >= used + attnb_bytes) ? (__bf16*)carve(attnb_bytes)
                                                  : (__bf16*)0;

  // conversions
  k_cvt_bf16<<<(BN * EE) / 256, 256, 0, stream>>>(x, xb, BN * EE);
  k_cvt_t<<<(EE * E2) / 256, 256, 0, stream>>>(Wq, wqt, EE, E2);
  k_cvt_t<<<(EE * E2) / 256, 256, 0, stream>>>(Wk, wkt, EE, E2);
  k_cvt_t<<<(EE * E2) / 256, 256, 0, stream>>>(Wv, wvt, EE, E2);
  k_cvt_t<<<(E2 * EE) / 256, 256, 0, stream>>>(Wo, wot, E2, EE);
  k_lambda<<<1, 64, 0, stream>>>(lq1, lk1, lq2, lk2, lam);

  // QKV projections
  dim3 gQKV(E2 / 64, BN / 128);
  k_gemm_qkv<<<gQKV, 256, 0, stream>>>(xb, wqt, Qb, 0);
  k_gemm_qkv<<<gQKV, 256, 0, stream>>>(xb, wkt, Kb, 0);
  k_gemm_qkv<<<gQKV, 256, 0, stream>>>(xb, wvt, Vtb, 1);

  // differential attention
  dim3 gAtt(BB * HH, NN / 128);
  k_stats<<<gAtt, 256, 0, stream>>>(Qb, Kb, mx1, sm1, mx2, sm2);
  k_attn<<<gAtt, 256, 0, stream>>>(Qb, Kb, mx1, sm1, mx2, sm2, lam, attn_out, attnb);
  k_pv<<<gAtt, 256, 0, stream>>>(attn_out, attnb, Vtb, Obuf);

  // GroupNorm
  k_gnstats<<<BB * HH, 256, 0, stream>>>(Obuf, mu, rsig);
  k_gnapply<<<(BB * HH * NN * D2) / 256, 256, 0, stream>>>(Obuf, mu, rsig, gamma, beta, Yb);

  // output projection
  dim3 gOut(EE / 64, BN / 128);
  k_out<<<gOut, 256, 0, stream>>>(Yb, wot, out);
}
